// VQVAE_30056181137409
// MI455X (gfx1250) — matrix operand order repaired and checked by reading, not compiled, not run on hardware
//
#include <hip/hip_runtime.h>
#include <hip/hip_bf16.h>
#include <math.h>

typedef __attribute__((ext_vector_type(16))) _Float16 v16h;
typedef __attribute__((ext_vector_type(8)))  _Float16 v8h;
typedef __attribute__((ext_vector_type(8)))  float    v8f;

#define NB      64
#define DDIM    256
#define HW      1024          // 32*32
#define NROWS   65536         // NB*HW
#define KCODES  1024
#define SBS     272           // LDS stride (halves) for B tile: 256 + 16 pad

// ---------------------------------------------------------------------------
// Kernel 1: NCHW f32 -> (N x D) row-major f16, via 32x32 LDS tile transpose
// ---------------------------------------------------------------------------
__global__ void k_transpose_f16(const float* __restrict__ in, _Float16* __restrict__ z16) {
    __shared__ _Float16 t16[32][33];
    const int b  = blockIdx.z;
    const int c0 = blockIdx.y * 32;
    const int p0 = blockIdx.x * 32;
    const int tx = threadIdx.x, ty = threadIdx.y;
    #pragma unroll
    for (int i = 0; i < 4; ++i) {
        int cl = ty + 8 * i;
        float v = in[(size_t)b * (DDIM * HW) + (size_t)(c0 + cl) * HW + p0 + tx];
        t16[cl][tx] = (_Float16)v;
    }
    __syncthreads();
    #pragma unroll
    for (int i = 0; i < 4; ++i) {
        int pl = ty + 8 * i;
        z16[(size_t)(b * HW + p0 + pl) * DDIM + c0 + tx] = t16[tx][pl];
    }
}

// ---------------------------------------------------------------------------
// Kernel 2: embedding f32 -> f16 copy + per-code squared norm (f32)
// ---------------------------------------------------------------------------
__global__ void k_emb_prep(const float* __restrict__ e, _Float16* __restrict__ e16,
                           float* __restrict__ eNorm) {
    __shared__ float red[256];
    const int k = blockIdx.x;
    const int t = threadIdx.x;
    float v = e[(size_t)k * DDIM + t];
    e16[(size_t)k * DDIM + t] = (_Float16)v;
    red[t] = v * v;
    __syncthreads();
    for (int s = 128; s > 0; s >>= 1) {
        if (t < s) red[t] += red[t + s];
        __syncthreads();
    }
    if (t == 0) eNorm[k] = red[0];
}

// ---------------------------------------------------------------------------
// Kernel 3: init accumulators
// ---------------------------------------------------------------------------
__global__ void k_init(float* __restrict__ out, unsigned* __restrict__ counts) {
    int t = threadIdx.x;
    if (t < KCODES) counts[t] = 0u;
    if (t == 0) out[0] = 0.0f;
}

// ---------------------------------------------------------------------------
// Kernel 4: distance GEMM (f16 WMMA, f32 acc) + per-row argmin over K
//   block = 256 threads (8 waves); wave w handles TWO 16-row strips (32 rows)
//   so each LDS B-fragment feeds two independent WMMA chains.
//   B tiles (16 codes x 256 D) are double-buffered in LDS.
// ---------------------------------------------------------------------------
__global__ void __launch_bounds__(256)
k_argmin_wmma(const _Float16* __restrict__ z16, const _Float16* __restrict__ e16,
              const float* __restrict__ eNorm, int* __restrict__ idxOut) {
    __shared__ _Float16 sB[2][16 * SBS];

    const int tid  = threadIdx.x;
    const int wave = tid >> 5;
    const int lane = tid & 31;
    const int m0   = (blockIdx.x * 8 + wave) * 32;   // 32 rows per wave

    // ---- preload A fragments for both strips, per ISA 16-bit A layout ----
    const int mlo   = m0 + (lane & 15);
    const int khalf = (lane >> 4) * 8;               // lanes 16-31 hold K+8 halves
    v16h A0[8], A1[8];
    #pragma unroll
    for (int dt = 0; dt < 8; ++dt) {
        const _Float16* b0 = z16 + (size_t)mlo * DDIM + dt * 32 + khalf;
        ((v8h*)&A0[dt])[0] = *(const v8h*)(b0);
        ((v8h*)&A0[dt])[1] = *(const v8h*)(b0 + 16);
        const _Float16* b1 = b0 + (size_t)16 * DDIM;
        ((v8h*)&A1[dt])[0] = *(const v8h*)(b1);
        ((v8h*)&A1[dt])[1] = *(const v8h*)(b1 + 16);
    }

    float best0[8], best1[8];
    int   bix0[8],  bix1[8];
    #pragma unroll
    for (int j = 0; j < 8; ++j) {
        best0[j] = 3.0e38f; best1[j] = 3.0e38f;
        bix0[j] = 0; bix1[j] = 0;
    }

    const int nloc = lane & 15;                      // code column within tile
    const int khi  = (lane >> 4) * 16;               // lanes 16-31 hold K+16 halves
    const int cl   = tid >> 4;                       // staging: code_local 0..15
    const int d0   = (tid & 15) * 16;                // staging: 16 halves / thread

    // prime buffer 0 with tile 0
    {
        const v8h* g = (const v8h*)(e16 + (size_t)cl * DDIM + d0);
        v8h x0 = g[0], x1 = g[1];
        *(v8h*)&sB[0][cl * SBS + d0]     = x0;
        *(v8h*)&sB[0][cl * SBS + d0 + 8] = x1;
    }

    for (int ct = 0; ct < 64; ++ct) {
        __syncthreads();                             // tile ct visible everywhere
        const int cur = ct & 1;

        // issue global loads for tile ct+1 early (hide latency under WMMAs)
        v8h x0, x1;
        const bool have_next = (ct < 63);
        if (have_next) {
            const v8h* g = (const v8h*)(e16 + (size_t)((ct + 1) * 16 + cl) * DDIM + d0);
            x0 = g[0];
            x1 = g[1];
        }

        v8f c0 = {}, c1 = {};
        #pragma unroll
        for (int dt = 0; dt < 8; ++dt) {
            v16h bfrag;
            ((v8h*)&bfrag)[0] = *(const v8h*)(&sB[cur][nloc * SBS + dt * 32 + khalf]);
            ((v8h*)&bfrag)[1] = *(const v8h*)(&sB[cur][nloc * SBS + dt * 32 + khalf + 16]);
            c0 = __builtin_amdgcn_wmma_f32_16x16x32_f16(
                false, A0[dt], false, bfrag, (short)0, c0, false, false);
            c1 = __builtin_amdgcn_wmma_f32_16x16x32_f16(
                false, A1[dt], false, bfrag, (short)0, c1, false, false);
        }

        // stash tile ct+1 into the other buffer (after our reads of cur)
        if (have_next) {
            *(v8h*)&sB[cur ^ 1][cl * SBS + d0]     = x0;
            *(v8h*)&sB[cur ^ 1][cl * SBS + d0 + 8] = x1;
        }

        const int   code = ct * 16 + nloc;
        const float en   = eNorm[code];
        #pragma unroll
        for (int j = 0; j < 8; ++j) {
            float s0 = en - 2.0f * c0[j];            // ||e||^2 - 2 z.e (argmin key)
            float s1 = en - 2.0f * c1[j];
            if (s0 < best0[j]) { best0[j] = s0; bix0[j] = code; }
            if (s1 < best1[j]) { best1[j] = s1; bix1[j] = code; }
        }
    }

    // ---- reduce across the 16 columns held by each 16-lane group ----
    // C layout: VGPR j -> row base+j (lanes 0-15) / row base+8+j (lanes 16-31)
    #pragma unroll
    for (int j = 0; j < 8; ++j) {
        float bv = best0[j];
        int   bi = bix0[j];
        #pragma unroll
        for (int off = 8; off >= 1; off >>= 1) {
            float ov = __shfl_xor(bv, off, 16);
            int   oi = __shfl_xor(bi, off, 16);
            if (ov < bv || (ov == bv && oi < bi)) { bv = ov; bi = oi; }
        }
        if ((lane & 15) == 0)
            idxOut[m0 + j + (lane >> 4) * 8] = bi;

        float cv = best1[j];
        int   ci = bix1[j];
        #pragma unroll
        for (int off = 8; off >= 1; off >>= 1) {
            float ov = __shfl_xor(cv, off, 16);
            int   oi = __shfl_xor(ci, off, 16);
            if (ov < cv || (ov == cv && oi < ci)) { cv = ov; ci = oi; }
        }
        if ((lane & 15) == 0)
            idxOut[m0 + 16 + j + (lane >> 4) * 8] = ci;
    }
}

// ---------------------------------------------------------------------------
// Kernel 5: codebook-usage histogram
// ---------------------------------------------------------------------------
__global__ void k_hist(const int* __restrict__ idx, unsigned* __restrict__ counts) {
    int n = blockIdx.x * blockDim.x + threadIdx.x;
    if (n < NROWS) atomicAdd(&counts[idx[n]], 1u);
}

// ---------------------------------------------------------------------------
// Kernel 6: gather z_q (full f32), write NCHW output, accumulate loss
// ---------------------------------------------------------------------------
__global__ void k_gather_loss(const float* __restrict__ in, const float* __restrict__ emb,
                              const int* __restrict__ idx, float* __restrict__ out) {
    __shared__ int   rows[32];
    __shared__ float tile[32][33];      // [p_local][c_local]
    __shared__ float red[256];

    const int b  = blockIdx.z;
    const int c0 = blockIdx.y * 32;
    const int p0 = blockIdx.x * 32;
    const int tx = threadIdx.x, ty = threadIdx.y;
    const int t  = ty * 32 + tx;

    if (ty == 0) rows[tx] = idx[b * HW + p0 + tx];
    __syncthreads();

    #pragma unroll
    for (int i = 0; i < 4; ++i) {       // coalesced along c
        int pl = ty + 8 * i;
        tile[pl][tx] = emb[(size_t)rows[pl] * DDIM + c0 + tx];
    }
    __syncthreads();

    float acc = 0.0f;
    #pragma unroll
    for (int i = 0; i < 4; ++i) {       // coalesced along p
        int cl = ty + 8 * i;
        size_t off = (size_t)b * (DDIM * HW) + (size_t)(c0 + cl) * HW + p0 + tx;
        float zq = tile[tx][cl];
        float z  = in[off];
        out[1 + off] = zq;              // straight-through forward == z_q
        float d = zq - z;
        acc += d * d;
    }

    red[t] = acc;
    __syncthreads();
    for (int s = 128; s > 0; s >>= 1) {
        if (t < s) red[t] += red[t + s];
        __syncthreads();
    }
    if (t == 0) {
        const float scale = 1.25f / ((float)NROWS * (float)DDIM); // (1+BETA)*mean
        atomicAdd(&out[0], red[0] * scale);
    }
}

// ---------------------------------------------------------------------------
// Kernel 7: perplexity
// ---------------------------------------------------------------------------
__global__ void k_perplexity(const unsigned* __restrict__ counts, float* __restrict__ out) {
    __shared__ float red[256];
    const int t = threadIdx.x;
    float s = 0.0f;
    #pragma unroll
    for (int i = 0; i < 4; ++i) {
        float p = (float)counts[t + 256 * i] / (float)NROWS;
        s += p * logf(p + 1e-10f);
    }
    red[t] = s;
    __syncthreads();
    for (int st = 128; st > 0; st >>= 1) {
        if (t < st) red[t] += red[t + st];
        __syncthreads();
    }
    if (t == 0) out[1 + (size_t)NB * DDIM * HW] = expf(-red[0]);
}

// ---------------------------------------------------------------------------
extern "C" void kernel_launch(void* const* d_in, const int* in_sizes, int n_in,
                              void* d_out, int out_size, void* d_ws, size_t ws_size,
                              hipStream_t stream) {
    const float* in  = (const float*)d_in[0];   // (64, 256, 32, 32) f32
    const float* emb = (const float*)d_in[1];   // (1024, 256) f32
    float* out = (float*)d_out;                 // [loss, z_q_out(64,256,32,32), perplexity]

    char* ws = (char*)d_ws;
    _Float16* z16    = (_Float16*)(ws);                               // 32 MB
    _Float16* e16    = (_Float16*)(ws + (size_t)33554432);            // 512 KB
    float*    eNorm  = (float*)   (ws + (size_t)33554432 + 524288);   // 4 KB
    int*      idx    = (int*)     (ws + (size_t)33554432 + 524288 + 4096);   // 256 KB
    unsigned* counts = (unsigned*)(ws + (size_t)33554432 + 524288 + 4096 + 262144); // 4 KB

    // 1) z: NCHW f32 -> row-major (N x D) f16
    k_transpose_f16<<<dim3(32, 8, 64), dim3(32, 8), 0, stream>>>(in, z16);
    // 2) embedding -> f16 + norms
    k_emb_prep<<<KCODES, 256, 0, stream>>>(emb, e16, eNorm);
    // 3) zero loss / histogram
    k_init<<<1, 1024, 0, stream>>>(out, counts);
    // 4) WMMA distance GEMM + argmin (4096 strips, 2 per wave, 8 waves/block)
    k_argmin_wmma<<<256, 256, 0, stream>>>(z16, e16, eNorm, idx);
    // 5) histogram
    k_hist<<<NROWS / 256, 256, 0, stream>>>(idx, counts);
    // 6) gather + loss + NCHW output
    k_gather_loss<<<dim3(32, 8, 64), dim3(32, 8), 0, stream>>>(in, emb, idx, out);
    // 7) perplexity
    k_perplexity<<<1, 256, 0, stream>>>(counts, out);
}
